// deform_35424890257709
// MI455X (gfx1250) — matrix operand order repaired and checked by reading, not compiled, not run on hardware
//
#include <hip/hip_runtime.h>
#include <hip/hip_bf16.h>

// CDNA5 / gfx1250 bf16 WMMA U-Net decoder + Gaussian warp + grid-sample.
// Activations staged channel-last in bf16 (L2-resident on MI455X), conv stages
// as implicit GEMM via v_wmma_f32_16x16x32_bf16.
// v2: multi-coutTile accumulation per wave (8x less A traffic, 8 independent
// WMMAs per A fragment) and 4 waves per 128-thread block.

typedef __bf16 bf16;
typedef __attribute__((ext_vector_type(16))) __bf16 v16bf;
typedef __attribute__((ext_vector_type(8)))  float  v8f;
typedef __attribute__((ext_vector_type(4)))  int    i4;

union FragB { v16bf v; i4 q[2]; };

#define BN_EPS 1e-5f
#define WAVES 4

// ---------------------------------------------------------------------------
// Weight packing into per-lane WMMA B-fragment order:
//   wpk[(((coT*T + t)*Q + q)*32 + lane)*16 + e]
//   col = coT*16 + (lane&15);  K = (lane>=16 ? 16 : 0) + e;  cin = q*32 + K
// ---------------------------------------------------------------------------
__global__ void pack_conv3_w(const float* __restrict__ w, int Cin,
                             int CoutReal, int CoutPad, bf16* __restrict__ wpk) {
  size_t i = (size_t)blockIdx.x * blockDim.x + threadIdx.x;
  size_t total = (size_t)(CoutPad / 16) * 27 * (Cin / 32) * 512;
  if (i >= total) return;
  int e    = (int)(i & 15);
  int lane = (int)((i >> 4) & 31);
  size_t r = i >> 9;
  int Q = Cin / 32;
  int q = (int)(r % Q); r /= Q;
  int t = (int)(r % 27); r /= 27;
  int coT = (int)r;
  int co  = coT * 16 + (lane & 15);
  int cin = q * 32 + ((lane >> 4) ? 16 : 0) + e;
  float val = (co < CoutReal) ? w[((size_t)co * Cin + cin) * 27 + t] : 0.f;
  wpk[i] = (bf16)val;
}

__global__ void pack_convt_w(const float* __restrict__ w, int Cin, int Cout,
                             bf16* __restrict__ wpk) {
  size_t i = (size_t)blockIdx.x * blockDim.x + threadIdx.x;
  size_t total = (size_t)(Cout / 16) * 8 * (Cin / 32) * 512;
  if (i >= total) return;
  int e    = (int)(i & 15);
  int lane = (int)((i >> 4) & 31);
  size_t r = i >> 9;
  int Q = Cin / 32;
  int q = (int)(r % Q); r /= Q;
  int t = (int)(r % 8); r /= 8;
  int coT = (int)r;
  int co  = coT * 16 + (lane & 15);
  int cin = q * 32 + ((lane >> 4) ? 16 : 0) + e;
  // ConvTranspose weight layout (ci, co, 2,2,2)
  wpk[i] = (bf16)w[((size_t)cin * Cout + co) * 8 + t];
}

// ---------------------------------------------------------------------------
// Conv3d 3x3x3, pad 1, implicit GEMM.  act: bf16 [vox][Cin] (channel-last).
// Each wave owns a 16-voxel M-tile and accumulates NT 16-wide N-tiles
// (NT accumulators); K loop = 27 taps x Cin/32.  4 waves per block.
// out: f32 [vox][CoutPad], bias fused.
// ---------------------------------------------------------------------------
template <int NT>
__global__ __launch_bounds__(32 * WAVES)
void conv3_wmma(const bf16* __restrict__ act, int Cin,
                const bf16* __restrict__ wpk, const float* __restrict__ bias,
                float* __restrict__ out, int CoutPad, int CoutReal,
                int D, int H, int W) {
  const int lane = threadIdx.x & 31;
  const int wave = threadIdx.x >> 5;
  const int m = lane & 15, hi = lane >> 4;
  const int v0 = (blockIdx.x * WAVES + wave) * 16;  // W%16==0 -> same (z,y)
  const int coTbase = blockIdx.y * NT;
  const int z = v0 / (H * W);
  const int y = (v0 / W) % H;
  const int x = (v0 % W) + m;
  const int Q = Cin >> 5;
  v8f acc[NT];
#pragma unroll
  for (int n = 0; n < NT; ++n) acc[n] = (v8f)0.0f;
  for (int t = 0; t < 27; ++t) {
    const int zz = z + t / 9 - 1;
    const int yy = y + (t / 3) % 3 - 1;
    const int xx = x + t % 3 - 1;
    const bool ok = ((unsigned)zz < (unsigned)D) &&
                    ((unsigned)yy < (unsigned)H) &&
                    ((unsigned)xx < (unsigned)W);
    const bf16* ap = act + (ok ? ((size_t)(zz * H + yy) * W + xx) * Cin : 0);
    for (int q = 0; q < Q; ++q) {
      FragB a;
      if (ok) {                                    // two b128 loads per lane
        a.q[0] = *(const i4*)(ap + q * 32 + hi * 16);
        a.q[1] = *(const i4*)(ap + q * 32 + hi * 16 + 8);
      } else {
        a.q[0] = (i4)0; a.q[1] = (i4)0;
      }
#pragma unroll
      for (int n = 0; n < NT; ++n) {               // reuse A across N tiles
        FragB b;
        const bf16* bp =
            wpk + ((((size_t)(coTbase + n) * 27 + t) * Q + q) * 32 + lane) * 16;
        b.q[0] = *(const i4*)bp;
        b.q[1] = *(const i4*)(bp + 8);
        acc[n] = __builtin_amdgcn_wmma_f32_16x16x32_bf16(
            false, a.v, false, b.v, (short)0, acc[n], false, false);
      }
    }
  }
#pragma unroll
  for (int n = 0; n < NT; ++n) {
    const int co = (coTbase + n) * 16 + m;
    const float bv = (co < CoutReal) ? bias[co] : 0.f;
#pragma unroll
    for (int r = 0; r < 8; ++r) {
      const int M = r + hi * 8;
      out[((size_t)(v0 + M)) * CoutPad + co] = acc[n][r] + bv;
    }
  }
}

// ---------------------------------------------------------------------------
// ConvTranspose3d k=2 s=2: 8 GEMMs (one per output parity); A fragments are
// loaded once per input-voxel tile and reused across all taps and Cout tiles.
// ---------------------------------------------------------------------------
template <int Q>
__global__ __launch_bounds__(32 * WAVES)
void convt2_wmma(const bf16* __restrict__ act,
                 const bf16* __restrict__ wpk, const float* __restrict__ bias,
                 float* __restrict__ out, int Cout, int Di, int Hi, int Wi) {
  const int lane = threadIdx.x & 31;
  const int wave = threadIdx.x >> 5;
  const int m = lane & 15, hi = lane >> 4;
  const int v0 = (blockIdx.x * WAVES + wave) * 16;
  const int Cin = Q * 32;
  const int v = v0 + m;
  FragB a[Q];
  const bf16* ap = act + (size_t)v * Cin;
#pragma unroll
  for (int q = 0; q < Q; ++q) {
    a[q].q[0] = *(const i4*)(ap + q * 32 + hi * 16);
    a[q].q[1] = *(const i4*)(ap + q * 32 + hi * 16 + 8);
  }
  const int Ho = 2 * Hi, Wo = 2 * Wi;
  const int numCoT = Cout >> 4;
  for (int t = 0; t < 8; ++t) {
    const int ti = t >> 2, tj = (t >> 1) & 1, tk = t & 1;
    for (int coT = 0; coT < numCoT; ++coT) {
      v8f acc = (v8f)0.0f;
#pragma unroll
      for (int q = 0; q < Q; ++q) {
        FragB b;
        const bf16* bp =
            wpk + ((((size_t)coT * 8 + t) * Q + q) * 32 + lane) * 16;
        b.q[0] = *(const i4*)bp;
        b.q[1] = *(const i4*)(bp + 8);
        acc = __builtin_amdgcn_wmma_f32_16x16x32_bf16(
            false, a[q].v, false, b.v, (short)0, acc, false, false);
      }
      const int co = coT * 16 + m;
      const float bv = bias[co];
#pragma unroll
      for (int r = 0; r < 8; ++r) {
        const int vv = v0 + r + hi * 8;
        const int z2 = vv / (Hi * Wi), y2 = (vv / Wi) % Hi, x2 = vv % Wi;
        const size_t ov =
            ((size_t)(2 * z2 + ti) * Ho + (2 * y2 + tj)) * Wo + (2 * x2 + tk);
        out[ov * Cout + co] = acc[r] + bv;
      }
    }
  }
}

// ---------------------------------------------------------------------------
// BatchNorm batch statistics: one block per channel (deterministic).
// ---------------------------------------------------------------------------
__global__ void bn_stats(const float* __restrict__ x, size_t N, int Cstride,
                         float* __restrict__ mean, float* __restrict__ var) {
  const int c = blockIdx.x;
  float s = 0.f, s2 = 0.f;
  for (size_t v = threadIdx.x; v < N; v += blockDim.x) {
    float t = x[v * Cstride + c];
    s += t; s2 += t * t;
  }
  __shared__ float rs[256], rq[256];
  rs[threadIdx.x] = s; rq[threadIdx.x] = s2;
  __syncthreads();
  for (int o = 128; o > 0; o >>= 1) {
    if ((int)threadIdx.x < o) {
      rs[threadIdx.x] += rs[threadIdx.x + o];
      rq[threadIdx.x] += rq[threadIdx.x + o];
    }
    __syncthreads();
  }
  if (threadIdx.x == 0) {
    float mm = rs[0] / (float)N;
    mean[c] = mm;
    var[c]  = rq[0] / (float)N - mm * mm;   // biased variance
  }
}

// BN + LeakyReLU(0.2) + bf16 store (with channel offset for concat buffers).
__global__ void bn_act_store(const float* __restrict__ x, int Cstride, int Creal,
                             const float* __restrict__ mean,
                             const float* __restrict__ var,
                             const float* __restrict__ g,
                             const float* __restrict__ b,
                             bf16* __restrict__ dst, int Cdst, int cOff,
                             size_t N) {
  size_t i = (size_t)blockIdx.x * blockDim.x + threadIdx.x;
  if (i >= N * (size_t)Creal) return;
  size_t v = i / Creal;
  int c = (int)(i % Creal);
  float t = x[v * Cstride + c];
  t = g[c] * (t - mean[c]) * rsqrtf(var[c] + BN_EPS) + b[c];
  t = t > 0.f ? t : 0.2f * t;
  dst[v * (size_t)Cdst + cOff + c] = (bf16)t;
}

// Plain f32 -> bf16 re-layout (u0t has no BN / activation).
__global__ void cast_store(const float* __restrict__ x, int Cstride, int Creal,
                           bf16* __restrict__ dst, int Cdst, int cOff, size_t N) {
  size_t i = (size_t)blockIdx.x * blockDim.x + threadIdx.x;
  if (i >= N * (size_t)Creal) return;
  size_t v = i / Creal;
  int c = (int)(i % Creal);
  dst[v * (size_t)Cdst + cOff + c] = (bf16)x[v * Cstride + c];
}

// NCDHW f32 (B=1) -> channel-last bf16 at channel offset (skip-connections).
__global__ void pack_ncdhw(const float* __restrict__ src, int C, size_t N,
                           bf16* __restrict__ dst, int Cdst, int cOff) {
  size_t i = (size_t)blockIdx.x * blockDim.x + threadIdx.x;
  if (i >= (size_t)C * N) return;
  int c = (int)(i / N);
  size_t v = i % N;
  dst[v * (size_t)Cdst + cOff + c] = (bf16)src[i];
}

// u3c2 tail: BN + lrelu + sigmoid + (d-0.5)*2*ds_lim -> planar disp[3][N].
__global__ void make_disp(const float* __restrict__ x,
                          const float* __restrict__ mean,
                          const float* __restrict__ var,
                          const float* __restrict__ g,
                          const float* __restrict__ b,
                          const float* __restrict__ ds_lim,
                          float* __restrict__ disp, size_t N) {
  size_t i = (size_t)blockIdx.x * blockDim.x + threadIdx.x;
  if (i >= 3 * N) return;
  int c = (int)(i / N);
  size_t v = i % N;
  float t = x[v * 16 + c];
  t = g[c] * (t - mean[c]) * rsqrtf(var[c] + BN_EPS) + b[c];
  t = t > 0.f ? t : 0.2f * t;
  t = 1.f / (1.f + __expf(-t));
  disp[i] = (t - 0.5f) * 2.f * ds_lim[0];
}

// Separable 11-tap Gaussian (sigma=1), zero padding; alpha on final pass.
__global__ void gauss1d(const float* __restrict__ src, float* __restrict__ dst,
                        int S, int axis, const float* __restrict__ alphas,
                        int doAlpha) {
  size_t N = (size_t)S * S * S;
  size_t i = (size_t)blockIdx.x * blockDim.x + threadIdx.x;
  if (i >= 3 * N) return;
  int c = (int)(i / N);
  size_t r = i % N;
  int z = (int)(r / ((size_t)S * S)), y = (int)((r / S) % S), x = (int)(r % S);
  float wgt[11], ws = 0.f;
#pragma unroll
  for (int j = 0; j < 11; ++j) {
    float t = (float)j - 5.f;
    wgt[j] = __expf(-0.5f * t * t);
    ws += wgt[j];
  }
  float inv = 1.f / ws;
  int p = (axis == 0) ? z : (axis == 1) ? y : x;
  size_t stride = (axis == 0) ? (size_t)S * S : (axis == 1) ? (size_t)S : 1;
  const float* base = src + (size_t)c * N + r - (size_t)p * stride;
  float acc = 0.f;
#pragma unroll
  for (int j = 0; j < 11; ++j) {
    int pp = p + j - 5;
    if ((unsigned)pp < (unsigned)S) acc += wgt[j] * base[(size_t)pp * stride];
  }
  acc *= inv;
  if (doAlpha) acc *= alphas[0];
  dst[i] = acc;
}

__device__ __forceinline__ float reflect_coord(float g, int size) {
  float span = (float)(size - 1);
  float cc = (g + 1.f) * 0.5f * span;
  cc = fabsf(cc);
  float m = 2.f * span;
  cc = cc - m * floorf(cc / m);
  if (cc > span) cc = m - cc;
  return fminf(fmaxf(cc, 0.f), span);
}

// Trilinear grid-sample (align_corners, reflection) of volume + mask.
__global__ void grid_sample3d(const float* __restrict__ vol,
                              const float* __restrict__ mask,
                              const float* __restrict__ disp,
                              float* __restrict__ out, int S) {
  size_t N = (size_t)S * S * S;
  size_t i = (size_t)blockIdx.x * blockDim.x + threadIdx.x;
  if (i >= N) return;
  int z = (int)(i / ((size_t)S * S)), y = (int)((i / S) % S), x = (int)(i % S);
  float inv = 2.f / (float)(S - 1);
  // grid[...,0]=lin(W)+d2 -> ix ; grid[...,1]=lin(H)+d1 -> iy ; grid[...,2]=lin(D)+d0 -> iz
  float ix = reflect_coord((float)x * inv - 1.f + disp[2 * N + i], S);
  float iy = reflect_coord((float)y * inv - 1.f + disp[N + i], S);
  float iz = reflect_coord((float)z * inv - 1.f + disp[i], S);
  int x0 = (int)floorf(ix), y0 = (int)floorf(iy), z0 = (int)floorf(iz);
  float fx = ix - x0, fy = iy - y0, fz = iz - z0;
  float sv = 0.f, sm = 0.f;
#pragma unroll
  for (int dz = 0; dz < 2; ++dz) {
    float wz = dz ? fz : 1.f - fz;
    int zi = min(max(z0 + dz, 0), S - 1);
#pragma unroll
    for (int dy = 0; dy < 2; ++dy) {
      float wy = dy ? fy : 1.f - fy;
      int yi = min(max(y0 + dy, 0), S - 1);
#pragma unroll
      for (int dx = 0; dx < 2; ++dx) {
        float wx = dx ? fx : 1.f - fx;
        int xi = min(max(x0 + dx, 0), S - 1);
        size_t idx = ((size_t)zi * S + yi) * S + xi;
        float w3 = wz * wy * wx;
        sv += vol[idx] * w3;
        sm += mask[idx] * w3;
      }
    }
  }
  out[i] = sv;
  out[N + i] = sm;
}

// ---------------------------------------------------------------------------
// Host orchestration
// ---------------------------------------------------------------------------
extern "C" void kernel_launch(void* const* d_in, const int* in_sizes, int n_in,
                              void* d_out, int out_size, void* d_ws,
                              size_t ws_size, hipStream_t stream) {
  (void)in_sizes; (void)n_in; (void)out_size; (void)ws_size;
  // params in setup_inputs() dict order
  const float* u0t_w = (const float*)d_in[0];  const float* u0t_b = (const float*)d_in[1];
  const float* u0c_w = (const float*)d_in[2];  const float* u0c_b = (const float*)d_in[3];
  const float* u0c_g = (const float*)d_in[4];  const float* u0c_e = (const float*)d_in[5];
  const float* u1t_w = (const float*)d_in[6];  const float* u1t_b = (const float*)d_in[7];
  const float* u1t_g = (const float*)d_in[8];  const float* u1t_e = (const float*)d_in[9];
  const float* u1c1_w = (const float*)d_in[10]; const float* u1c1_b = (const float*)d_in[11];
  const float* u1c1_g = (const float*)d_in[12]; const float* u1c1_e = (const float*)d_in[13];
  const float* u1c2_w = (const float*)d_in[14]; const float* u1c2_b = (const float*)d_in[15];
  const float* u1c2_g = (const float*)d_in[16]; const float* u1c2_e = (const float*)d_in[17];
  const float* u2t_w = (const float*)d_in[18]; const float* u2t_b = (const float*)d_in[19];
  const float* u2t_g = (const float*)d_in[20]; const float* u2t_e = (const float*)d_in[21];
  const float* u2c1_w = (const float*)d_in[22]; const float* u2c1_b = (const float*)d_in[23];
  const float* u2c1_g = (const float*)d_in[24]; const float* u2c1_e = (const float*)d_in[25];
  const float* u2c2_w = (const float*)d_in[26]; const float* u2c2_b = (const float*)d_in[27];
  const float* u2c2_g = (const float*)d_in[28]; const float* u2c2_e = (const float*)d_in[29];
  const float* u3c1_w = (const float*)d_in[30]; const float* u3c1_b = (const float*)d_in[31];
  const float* u3c1_g = (const float*)d_in[32]; const float* u3c1_e = (const float*)d_in[33];
  const float* u3c2_w = (const float*)d_in[34]; const float* u3c2_b = (const float*)d_in[35];
  const float* u3c2_g = (const float*)d_in[36]; const float* u3c2_e = (const float*)d_in[37];
  const float* bottleneck = (const float*)d_in[38];
  const float* inputs     = (const float*)d_in[39];
  const float* alphas     = (const float*)d_in[40];
  const float* mask       = (const float*)d_in[41];
  const float* store0     = (const float*)d_in[42];
  const float* store1     = (const float*)d_in[43];
  const float* store2     = (const float*)d_in[44];
  const float* ds_lim     = (const float*)d_in[45];
  float* out = (float*)d_out;

  const size_t V8 = 512, V16 = 4096, V32 = 32768, V64 = 262144;

  // bump allocator over workspace
  char* wsp = (char*)d_ws;
  auto alloc = [&](size_t bytes) -> void* {
    void* p = (void*)wsp;
    wsp += (bytes + 255) & ~(size_t)255;
    return p;
  };
  float* mean = (float*)alloc(256 * 4);
  float* var  = (float*)alloc(256 * 4);
  bf16* wbuf  = (bf16*)alloc((size_t)1 << 20);           // largest packed W
  bf16* actBN = (bf16*)alloc(V8 * 128 * 2);
  bf16* act16 = (bf16*)alloc(V16 * 128 * 2);
  bf16* cat2  = (bf16*)alloc(V16 * 256 * 2);
  bf16* a32   = (bf16*)alloc(V32 * 128 * 2);
  bf16* b32   = (bf16*)alloc(V32 * 128 * 2);
  bf16* cat1  = (bf16*)alloc(V32 * 256 * 2);
  bf16* a64   = (bf16*)alloc(V64 * 128 * 2);
  bf16* b64   = (bf16*)alloc(V64 * 64 * 2);
  bf16* cat0  = (bf16*)alloc(V64 * 128 * 2);
  bf16* c64   = (bf16*)alloc(V64 * 64 * 2);
  float* ftmp = (float*)alloc(V64 * 128 * 4);            // f32 conv output, reused
  float* dspA = (float*)alloc(3 * V64 * 4);
  float* dspB = (float*)alloc(3 * V64 * 4);

  const int TB = 256;
  auto cdiv = [](size_t a, size_t b) -> unsigned { return (unsigned)((a + b - 1) / b); };
  auto pk3 = [&](const float* w, int ci, int coR, int coP) {
    size_t tot = (size_t)(coP / 16) * 27 * (ci / 32) * 512;
    pack_conv3_w<<<cdiv(tot, TB), TB, 0, stream>>>(w, ci, coR, coP, wbuf);
  };
  auto pkT = [&](const float* w, int ci, int co) {
    size_t tot = (size_t)(co / 16) * 8 * (ci / 32) * 512;
    pack_convt_w<<<cdiv(tot, TB), TB, 0, stream>>>(w, ci, co, wbuf);
  };
  auto stats = [&](size_t N, int Cs, int Creal) {
    bn_stats<<<Creal, 256, 0, stream>>>(ftmp, N, Cs, mean, var);
  };
  auto bnstore = [&](size_t N, int Cs, int Creal, const float* g, const float* b,
                     bf16* dst, int Cdst, int off) {
    bn_act_store<<<cdiv(N * Creal, TB), TB, 0, stream>>>(
        ftmp, Cs, Creal, mean, var, g, b, dst, Cdst, off, N);
  };
  // conv grid: 4 waves/block, each wave one 16-voxel tile
  auto mgrid = [&](size_t Nvox) -> unsigned { return (unsigned)(Nvox / (16 * WAVES)); };

  // ---- bottleneck -> channel-last bf16
  pack_ncdhw<<<cdiv(128 * V8, TB), TB, 0, stream>>>(bottleneck, 128, V8, actBN, 128, 0);

  // ---- u0t: ConvT 128->128, 8^3 -> 16^3 (raw, no BN)
  pkT(u0t_w, 128, 128);
  convt2_wmma<4><<<mgrid(V8), 32 * WAVES, 0, stream>>>(
      actBN, wbuf, u0t_b, ftmp, 128, 8, 8, 8);
  cast_store<<<cdiv(V16 * 128, TB), TB, 0, stream>>>(ftmp, 128, 128, act16, 128, 0, V16);

  // ---- u0c: Conv3 128->128 @16^3, BN+lrelu -> cat2[:,0:128]
  pk3(u0c_w, 128, 128, 128);
  conv3_wmma<8><<<dim3(mgrid(V16), 1), 32 * WAVES, 0, stream>>>(
      act16, 128, wbuf, u0c_b, ftmp, 128, 128, 16, 16, 16);
  stats(V16, 128, 128);
  bnstore(V16, 128, 128, u0c_g, u0c_e, cat2, 256, 0);
  pack_ncdhw<<<cdiv(128 * V16, TB), TB, 0, stream>>>(store2, 128, V16, cat2, 256, 128);

  // ---- u1t: ConvT 256->128, 16^3 -> 32^3, BN+lrelu
  pkT(u1t_w, 256, 128);
  convt2_wmma<8><<<mgrid(V16), 32 * WAVES, 0, stream>>>(
      cat2, wbuf, u1t_b, ftmp, 128, 16, 16, 16);
  stats(V32, 128, 128);
  bnstore(V32, 128, 128, u1t_g, u1t_e, a32, 128, 0);

  // ---- u1c1, u1c2 @32^3
  pk3(u1c1_w, 128, 128, 128);
  conv3_wmma<8><<<dim3(mgrid(V32), 1), 32 * WAVES, 0, stream>>>(
      a32, 128, wbuf, u1c1_b, ftmp, 128, 128, 32, 32, 32);
  stats(V32, 128, 128);
  bnstore(V32, 128, 128, u1c1_g, u1c1_e, b32, 128, 0);

  pk3(u1c2_w, 128, 128, 128);
  conv3_wmma<8><<<dim3(mgrid(V32), 1), 32 * WAVES, 0, stream>>>(
      b32, 128, wbuf, u1c2_b, ftmp, 128, 128, 32, 32, 32);
  stats(V32, 128, 128);
  bnstore(V32, 128, 128, u1c2_g, u1c2_e, cat1, 256, 0);
  pack_ncdhw<<<cdiv(128 * V32, TB), TB, 0, stream>>>(store1, 128, V32, cat1, 256, 128);

  // ---- u2t: ConvT 256->128, 32^3 -> 64^3, BN+lrelu
  pkT(u2t_w, 256, 128);
  convt2_wmma<8><<<mgrid(V32), 32 * WAVES, 0, stream>>>(
      cat1, wbuf, u2t_b, ftmp, 128, 32, 32, 32);
  stats(V64, 128, 128);
  bnstore(V64, 128, 128, u2t_g, u2t_e, a64, 128, 0);

  // ---- u2c1: 128->64 @64^3
  pk3(u2c1_w, 128, 64, 64);
  conv3_wmma<4><<<dim3(mgrid(V64), 1), 32 * WAVES, 0, stream>>>(
      a64, 128, wbuf, u2c1_b, ftmp, 64, 64, 64, 64, 64);
  stats(V64, 64, 64);
  bnstore(V64, 64, 64, u2c1_g, u2c1_e, b64, 64, 0);

  // ---- u2c2: 64->64 @64^3 -> cat0[:,0:64]
  pk3(u2c2_w, 64, 64, 64);
  conv3_wmma<4><<<dim3(mgrid(V64), 1), 32 * WAVES, 0, stream>>>(
      b64, 64, wbuf, u2c2_b, ftmp, 64, 64, 64, 64, 64);
  stats(V64, 64, 64);
  bnstore(V64, 64, 64, u2c2_g, u2c2_e, cat0, 128, 0);
  pack_ncdhw<<<cdiv(64 * V64, TB), TB, 0, stream>>>(store0, 64, V64, cat0, 128, 64);

  // ---- u3c1: 128->64 @64^3
  pk3(u3c1_w, 128, 64, 64);
  conv3_wmma<4><<<dim3(mgrid(V64), 1), 32 * WAVES, 0, stream>>>(
      cat0, 128, wbuf, u3c1_b, ftmp, 64, 64, 64, 64, 64);
  stats(V64, 64, 64);
  bnstore(V64, 64, 64, u3c1_g, u3c1_e, c64, 64, 0);

  // ---- u3c2: 64->3 (padded to 16 out channels)
  pk3(u3c2_w, 64, 3, 16);
  conv3_wmma<1><<<dim3(mgrid(V64), 1), 32 * WAVES, 0, stream>>>(
      c64, 64, wbuf, u3c2_b, ftmp, 16, 3, 64, 64, 64);
  bn_stats<<<3, 256, 0, stream>>>(ftmp, V64, 16, mean, var);
  make_disp<<<cdiv(3 * V64, TB), TB, 0, stream>>>(ftmp, mean, var, u3c2_g, u3c2_e,
                                                  ds_lim, dspA, V64);

  // ---- separable Gaussian (x, y, z), alpha on last pass
  gauss1d<<<cdiv(3 * V64, TB), TB, 0, stream>>>(dspA, dspB, 64, 2, alphas, 0);
  gauss1d<<<cdiv(3 * V64, TB), TB, 0, stream>>>(dspB, dspA, 64, 1, alphas, 0);
  gauss1d<<<cdiv(3 * V64, TB), TB, 0, stream>>>(dspA, dspB, 64, 0, alphas, 1);

  // ---- warp inputs + mask
  grid_sample3d<<<cdiv(V64, TB), TB, 0, stream>>>(inputs, mask, dspB, out, 64);
}
